// SelfAttention_44100724196069
// MI455X (gfx1250) — matrix operand order repaired and checked by reading, not compiled, not run on hardware
//
#include <hip/hip_runtime.h>
#include <hip/hip_bf16.h>

typedef __attribute__((ext_vector_type(16))) __bf16 v16bf;
typedef __attribute__((ext_vector_type(8)))  __bf16 v8bf;
typedef __attribute__((ext_vector_type(8)))  float  v8f;
typedef __attribute__((ext_vector_type(4)))  float  v4f;
typedef __attribute__((ext_vector_type(4)))  unsigned int u32x4;
typedef __attribute__((ext_vector_type(8)))  unsigned int u32x8;

#define C_DIM 512
#define T_DIM 1024
#define B_DIM 32
#define ROWS  (B_DIM * T_DIM)          // 32768
#define SCALE 22.627416997969522f      // C ** 0.5 (faithful to reference)

// Padded LDS K-tile: 32 rows x (512 bf16 + 16B pad) -> row stride 1040 B
// (260 dwords == 4 banks mod 64, so 16-lane b128 reads are conflict-free).
#define KROW_BYTES 1040

// ---------------------------------------------------------------------------
// Kernel 1: q/k/v projections.  out = x @ W^T, bf16 results to workspace.
// One wave -> 16(M) x 64(N) tile, K-loop over C in steps of 32.
// proj 0 -> q (row major), 1 -> k (row major), 2 -> v stored transposed
// per-batch as vT[b][d][t] so the attention PV B-operand is contiguous.
// ---------------------------------------------------------------------------
__global__ __launch_bounds__(256) void qkv_proj_kernel(
    const float* __restrict__ x, const float* __restrict__ Wk,
    const float* __restrict__ Wq, const float* __restrict__ Wv,
    __bf16* __restrict__ qb, __bf16* __restrict__ kb, __bf16* __restrict__ vtb)
{
    const int lane  = threadIdx.x & 31;
    const int wid   = blockIdx.x * (blockDim.x >> 5) + (threadIdx.x >> 5);
    const int n4    = wid & 7;           // which 64-wide N strip (512/64 = 8)
    const int rest  = wid >> 3;
    const int mtile = rest & 2047;       // 32768/16 row tiles
    const int proj  = rest >> 11;        // 0..2
    if (proj >= 3) return;

    const int m0  = mtile << 4;
    const int n0  = n4 << 6;
    const int hi  = lane >> 4;
    const int l15 = lane & 15;

    const float* W = (proj == 0) ? Wq : (proj == 1) ? Wk : Wv;

    v8f acc[4] = {};
    // A row pointer: lane holds row m0+l15; halves are K = 16*hi + 0..15 (contiguous)
    const float* arow = x + (size_t)(m0 + l15) * C_DIM + hi * 16;
    // B: lane holds column n (= W row n); halves are K = 16*hi + 0..15 (contiguous)
    const float* wrow[4];
#pragma unroll
    for (int j = 0; j < 4; ++j)
        wrow[j] = W + (size_t)(n0 + 16 * j + l15) * C_DIM + hi * 16;

    for (int k0 = 0; k0 < C_DIM; k0 += 32) {
        v16bf a;
        {
            const v4f f0 = *(const v4f*)(arow + k0);
            const v4f f1 = *(const v4f*)(arow + k0 + 4);
            const v4f f2 = *(const v4f*)(arow + k0 + 8);
            const v4f f3 = *(const v4f*)(arow + k0 + 12);
#pragma unroll
            for (int i = 0; i < 4; ++i) {
                a[i]      = (__bf16)f0[i];
                a[4 + i]  = (__bf16)f1[i];
                a[8 + i]  = (__bf16)f2[i];
                a[12 + i] = (__bf16)f3[i];
            }
        }
#pragma unroll
        for (int j = 0; j < 4; ++j) {
            v16bf bm;
            const v4f g0 = *(const v4f*)(wrow[j] + k0);
            const v4f g1 = *(const v4f*)(wrow[j] + k0 + 4);
            const v4f g2 = *(const v4f*)(wrow[j] + k0 + 8);
            const v4f g3 = *(const v4f*)(wrow[j] + k0 + 12);
#pragma unroll
            for (int i = 0; i < 4; ++i) {
                bm[i]      = (__bf16)g0[i];
                bm[4 + i]  = (__bf16)g1[i];
                bm[8 + i]  = (__bf16)g2[i];
                bm[12 + i] = (__bf16)g3[i];
            }
            acc[j] = __builtin_amdgcn_wmma_f32_16x16x32_bf16(
                false, a, false, bm, (short)0, acc[j], false, false);
        }
    }

    if (proj != 2) {
        __bf16* dst = (proj == 0) ? qb : kb;
#pragma unroll
        for (int j = 0; j < 4; ++j) {
            const int n = n0 + 16 * j + l15;       // D: lane -> column n
#pragma unroll
            for (int r = 0; r < 8; ++r)            // vgpr r -> row m0 + r + 8*hi
                dst[(size_t)(m0 + r + 8 * hi) * C_DIM + n] = (__bf16)acc[j][r];
        }
    } else {
        // v transposed: vT[b][d][t]; the 8 rows of a D column are contiguous in t
        const int b = m0 >> 10;
        const int t = (m0 & 1023) + 8 * hi;
#pragma unroll
        for (int j = 0; j < 4; ++j) {
            const int d = n0 + 16 * j + l15;
            v8bf st;
#pragma unroll
            for (int r = 0; r < 8; ++r) st[r] = (__bf16)acc[j][r];
            *(v8bf*)(vtb + ((size_t)b * C_DIM + d) * T_DIM + t) = st;
        }
    }
}

// ---------------------------------------------------------------------------
// TDM: DMA one 32x512 bf16 K-tile (rows contiguous, row stride C_DIM) into
// LDS with 16B padding per 1024B row (D# pad_interval=256 dw, pad_amount=4 dw).
// 2-D tensor -> descriptor groups 0 and 1 only.
// ---------------------------------------------------------------------------
__device__ __forceinline__ void tdm_load_k_tile(const __bf16* gptr,
                                                unsigned lds_off)
{
    const unsigned long long ga = (unsigned long long)gptr;
    u32x4 g0;
    g0[0] = 1u;                                   // count = 1 (valid user D#)
    g0[1] = lds_off;                              // lds_addr
    g0[2] = (unsigned)ga;                         // global_addr[31:0]
    g0[3] = (unsigned)(ga >> 32) | 0x80000000u;   // global_addr[56:32] | type=2
    u32x8 g1;
    g1[0] = 0x07D10000u;      // data_size=2B | pad_en | interval=256dw | amt=4dw
    g1[1] = 512u << 16;       // tensor_dim0 = 512 (row length, elements)
    g1[2] = 1024u << 16;      // tensor_dim1 = 1024 (rows)
    g1[3] = 512u << 16;       // tile_dim0   = 512
    g1[4] = 32u;              // tile_dim1   = 32, tile_dim2 = 0
    g1[5] = 512u;             // tensor_dim0_stride = 512
    g1[6] = 0u;
    g1[7] = 0u;
    asm volatile("tensor_load_to_lds %0, %1" :: "s"(g0), "s"(g1) : "memory");
}

// ---------------------------------------------------------------------------
// Kernel 2: causal flash attention with TDM-staged K tiles.
// Block = 8 waves = 4 consecutive query tiles (same batch) x 2 d-halves.
// Wave 0 double-buffers the shared 32x512 K tile via tensor_load_to_lds
// (TENSORcnt); all waves read B-fragments from LDS.  Uniform (max) causal
// trip count per block: fully-masked steps contribute exactly zero.
// ---------------------------------------------------------------------------
__global__ __launch_bounds__(256) void attn_kernel(
    const __bf16* __restrict__ qb, const __bf16* __restrict__ kb,
    const __bf16* __restrict__ vtb, float* __restrict__ out)
{
    __shared__ __align__(128) char  ldsK[2][32 * KROW_BYTES];   // 2 x 32.5 KB
    __shared__ __align__(32) __bf16 ldsP[8][16][32];            // per-wave P tile

    const int lane  = threadIdx.x & 31;
    const int lwave = threadIdx.x >> 5;
    const int pair  = lwave >> 1;               // query tile within block (0..3)
    const int half  = lwave & 1;
    const int qt    = blockIdx.x * 4 + pair;    // 0..2047 (4 tiles per block)
    const int b     = qt >> 6;
    const int t0    = (qt & 63) << 4;
    const int hi    = lane >> 4;
    const int l15   = lane & 15;
    const int dbase = half << 8;                // 0 or 256

    const size_t rowbase = (size_t)b * T_DIM;
    const __bf16* qrow = qb + (rowbase + t0 + l15) * C_DIM;    // A-layout row
    const __bf16* kbb  = kb + rowbase * C_DIM;                 // batch K base
    const __bf16* vtbb = vtb + (size_t)b * C_DIM * T_DIM;

    v8f acc[16] = {};
    float mrow[8], lrow[8];
#pragma unroll
    for (int r = 0; r < 8; ++r) { mrow[r] = -1e30f; lrow[r] = 0.f; }

    // Uniform trip count over the block's last query tile (t0 of pair 3).
    const int t0_last = ((blockIdx.x * 4 + 3) & 63) << 4;
    const int ns = ((t0_last + 15) >> 5) + 1;

    if (lwave == 0)                                   // prime buffer 0
        tdm_load_k_tile(kbb, (unsigned)(size_t)&ldsK[0][0]);

    for (int si = 0; si < ns; ++si) {
        const int s0  = si << 5;
        const int cur = si & 1;
        if (lwave == 0) {
            if (si + 1 < ns) {                        // prefetch next K tile
                tdm_load_k_tile(kbb + (size_t)(si + 1) * 32 * C_DIM,
                                (unsigned)(size_t)&ldsK[cur ^ 1][0]);
                __builtin_amdgcn_s_wait_tensorcnt((short)1);  // buf[cur] done
            } else {
                __builtin_amdgcn_s_wait_tensorcnt((short)0);
            }
        }
        __syncthreads();                              // publish buf[cur]

        // ---- S = q @ k^T : 16x32 tile, K = 512; B-fragments from LDS ----
        v8f sacc0 = {}, sacc1 = {};
        for (int kk = 0; kk < C_DIM; kk += 32) {
            v16bf aq;
            {
                const v8bf q0 = *(const v8bf*)(qrow + kk + hi * 16);
                const v8bf q1 = *(const v8bf*)(qrow + kk + hi * 16 + 8);
#pragma unroll
                for (int i = 0; i < 8; ++i) { aq[i] = q0[i]; aq[8 + i] = q1[i]; }
            }
            v16bf bk0, bk1;
            {
                const int cb = (kk + hi * 16) * 2;
                const v8bf a0 = *(const v8bf*)&ldsK[cur][l15 * KROW_BYTES + cb];
                const v8bf a1 = *(const v8bf*)&ldsK[cur][l15 * KROW_BYTES + cb + 16];
                const v8bf b0 = *(const v8bf*)&ldsK[cur][(16 + l15) * KROW_BYTES + cb];
                const v8bf b1 = *(const v8bf*)&ldsK[cur][(16 + l15) * KROW_BYTES + cb + 16];
#pragma unroll
                for (int i = 0; i < 8; ++i) {
                    bk0[i] = a0[i]; bk0[8 + i] = a1[i];
                    bk1[i] = b0[i]; bk1[8 + i] = b1[i];
                }
            }
            sacc0 = __builtin_amdgcn_wmma_f32_16x16x32_bf16(
                false, aq, false, bk0, (short)0, sacc0, false, false);
            sacc1 = __builtin_amdgcn_wmma_f32_16x16x32_bf16(
                false, aq, false, bk1, (short)0, sacc1, false, false);
        }

        // ---- online softmax (D-layout: lane=col l15, vgpr r = row r+8*hi) ----
        float p0[8], p1[8], alpha[8];
#pragma unroll
        for (int r = 0; r < 8; ++r) {
            const int trow = t0 + r + 8 * hi;
            float v0 = sacc0[r] * SCALE;
            float v1 = sacc1[r] * SCALE;
            if (s0 + l15      > trow) v0 = -1e30f;   // causal mask
            if (s0 + 16 + l15 > trow) v1 = -1e30f;
            p0[r] = v0; p1[r] = v1;
        }
#pragma unroll
        for (int r = 0; r < 8; ++r) {
            float mx = fmaxf(p0[r], p1[r]);
#pragma unroll
            for (int ofs = 1; ofs < 16; ofs <<= 1)
                mx = fmaxf(mx, __shfl_xor(mx, ofs, 32));
            const float mnew = fmaxf(mrow[r], mx);
            const float a    = __expf(mrow[r] - mnew);
            alpha[r] = a;
            mrow[r]  = mnew;
            const float e0 = __expf(p0[r] - mnew);
            const float e1 = __expf(p1[r] - mnew);
            p0[r] = e0; p1[r] = e1;
            float rs = e0 + e1;
#pragma unroll
            for (int ofs = 1; ofs < 16; ofs <<= 1)
                rs += __shfl_xor(rs, ofs, 32);
            lrow[r] = lrow[r] * a + rs;
        }
#pragma unroll
        for (int j = 0; j < 16; ++j)
#pragma unroll
            for (int r = 0; r < 8; ++r)
                acc[j][r] *= alpha[r];

        // ---- transpose P (D-layout -> A-layout) through wave-private LDS ----
#pragma unroll
        for (int r = 0; r < 8; ++r) {
            ldsP[lwave][r + 8 * hi][l15]      = (__bf16)p0[r];
            ldsP[lwave][r + 8 * hi][l15 + 16] = (__bf16)p1[r];
        }
        asm volatile("s_wait_dscnt 0" ::: "memory");
        v16bf ap;
        {
            const v8bf w0 = *(const v8bf*)&ldsP[lwave][l15][hi * 16];
            const v8bf w1 = *(const v8bf*)&ldsP[lwave][l15][hi * 16 + 8];
#pragma unroll
            for (int i = 0; i < 8; ++i) { ap[i] = w0[i]; ap[8 + i] = w1[i]; }
        }

        // ---- out += P @ V : B operand from vT is one contiguous 32B row ----
        if (si + 1 < ns)    // prefetch next V columns (global_prefetch_b8)
            __builtin_prefetch(vtbb + (size_t)(dbase + l15) * T_DIM + s0 + 32, 0, 0);
#pragma unroll
        for (int j = 0; j < 16; ++j) {
            const int d = dbase + 16 * j + l15;
            const v16bf bv =
                *(const v16bf*)(vtbb + (size_t)d * T_DIM + s0 + hi * 16);
            acc[j] = __builtin_amdgcn_wmma_f32_16x16x32_bf16(
                false, ap, false, bv, (short)0, acc[j], false, false);
        }

        __syncthreads();   // all reads of buf[cur] done before it is re-DMAed
    }

    // ---- epilogue: divide by softmax denominator, store fp32 ----
#pragma unroll
    for (int r = 0; r < 8; ++r) lrow[r] = 1.0f / lrow[r];
    float* orow = out + (rowbase + t0) * C_DIM;
#pragma unroll
    for (int j = 0; j < 16; ++j) {
        const int d = dbase + 16 * j + l15;
#pragma unroll
        for (int r = 0; r < 8; ++r)
            orow[(size_t)(r + 8 * hi) * C_DIM + d] = acc[j][r] * lrow[r];
    }
}

// ---------------------------------------------------------------------------
extern "C" void kernel_launch(void* const* d_in, const int* in_sizes, int n_in,
                              void* d_out, int out_size, void* d_ws, size_t ws_size,
                              hipStream_t stream) {
    (void)in_sizes; (void)n_in; (void)out_size;
    const float* x  = (const float*)d_in[0];
    const float* Wk = (const float*)d_in[1];
    const float* Wq = (const float*)d_in[2];
    const float* Wv = (const float*)d_in[3];
    float* out = (float*)d_out;

    const size_t qkv_elems = (size_t)ROWS * C_DIM;          // 16M bf16 each
    if (ws_size < 3 * qkv_elems * sizeof(__bf16)) return;   // need ~96 MB
    __bf16* qb  = (__bf16*)d_ws;
    __bf16* kb  = (__bf16*)((char*)d_ws + qkv_elems * sizeof(__bf16));
    __bf16* vtb = (__bf16*)((char*)d_ws + 2 * qkv_elems * sizeof(__bf16));

    // 3 proj * 2048 mtiles * 8 nstrips = 49152 waves / 8 waves per block
    qkv_proj_kernel<<<6144, 256, 0, stream>>>(x, Wk, Wq, Wv, qb, kb, vtb);
    // 2048 query tiles / 4 per block, 8 waves (256 threads) per block
    attn_kernel<<<512, 256, 0, stream>>>(qb, kb, vtb, out);
}
